// SelfAttention_57002805952906
// MI455X (gfx1250) — matrix operand order repaired and checked by reading, not compiled, not run on hardware
//
#include <hip/hip_runtime.h>
#include <math.h>

// Fused causal self-attention for MI455X (gfx1250, wave32, WMMA).
// Pipeline: fp32 -> bf16 convert, WMMA QKV projection, flash-attention
// with v_wmma_f32_16x16x32_bf16 for both QK^T and P*V.

#define DI __device__ __forceinline__

static constexpr int B = 8, T = 2048, E = 128, H = 256;
static constexpr int M = B * T;

typedef __attribute__((ext_vector_type(16))) __bf16          v16bf;
typedef __attribute__((ext_vector_type(16))) unsigned short  v16u;
typedef __attribute__((ext_vector_type(8)))  unsigned short  v8u;
typedef __attribute__((ext_vector_type(8)))  float           v8f;

union U16x16 { v16u v; v8u h[2]; };

DI v8f wmma_bf16(v16u a, v16u b, v8f c) {
  return __builtin_amdgcn_wmma_f32_16x16x32_bf16(
      false, __builtin_bit_cast(v16bf, a),
      false, __builtin_bit_cast(v16bf, b),
      (short)0, c, false, false);
}

DI unsigned short f2bf(float f) {  // round-to-nearest-even f32 -> bf16
  unsigned int u = __float_as_uint(f);
  u += 0x7FFFu + ((u >> 16) & 1u);
  return (unsigned short)(u >> 16);
}

// ---------------- conversion kernels ----------------

__global__ __launch_bounds__(256) void k_cvt_x(const float* __restrict__ x,
                                               unsigned short* __restrict__ xb,
                                               int n) {
  int i = blockIdx.x * 256 + threadIdx.x;
  if (i < n) xb[i] = f2bf(x[i]);
}

// wt[mat][h][e] = W_mat[e][h]  (bf16, E-contiguous for B-fragments)
__global__ __launch_bounds__(256) void k_cvt_w(const float* __restrict__ Wq,
                                               const float* __restrict__ Wk,
                                               const float* __restrict__ Wv,
                                               unsigned short* __restrict__ wt) {
  int i = blockIdx.x * 256 + threadIdx.x;
  if (i >= 3 * H * E) return;
  int mat = i / (H * E), rem = i % (H * E);
  int h = rem / E, e = rem % E;
  const float* W = (mat == 0) ? Wq : (mat == 1) ? Wk : Wv;
  wt[i] = f2bf(W[e * H + h]);
}

// ---------------- QKV projection (WMMA) ----------------
// One wave computes a 16-row x 64-col tile of one projection.
// qb,kb: [M][H] bf16 ; vt: [B][H][T] bf16 (transposed for PV B-fragments)

__global__ __launch_bounds__(32) void k_proj(const unsigned short* __restrict__ xb,
                                             const unsigned short* __restrict__ wt,
                                             const float* __restrict__ bq,
                                             const float* __restrict__ bk,
                                             const float* __restrict__ bv,
                                             unsigned short* __restrict__ qb,
                                             unsigned short* __restrict__ kb,
                                             unsigned short* __restrict__ vt) {
  const int lane = threadIdx.x;
  const int ln = lane & 15;
  const int hf = lane >> 4;
  const int g0 = blockIdx.x * 16;          // row tile base over M
  const int mat = blockIdx.y >> 2;         // 0=q 1=k 2=v
  const int n0 = (blockIdx.y & 3) * 64;    // col group base

  // A fragments: x rows, K = E = 128 in 4 chunks of 32
  U16x16 A[4];
  {
    const unsigned short* xr = xb + (size_t)(g0 + ln) * E;
#pragma unroll
    for (int c = 0; c < 4; ++c) {
      A[c].h[0] = *(const v8u*)(xr + c * 32 + hf * 16);
      A[c].h[1] = *(const v8u*)(xr + c * 32 + hf * 16 + 8);
    }
  }

  const v8f zero = {0.f, 0.f, 0.f, 0.f, 0.f, 0.f, 0.f, 0.f};
  v8f acc[4];
#pragma unroll
  for (int t = 0; t < 4; ++t) acc[t] = zero;

#pragma unroll
  for (int t = 0; t < 4; ++t) {
    const unsigned short* wr =
        wt + ((size_t)(mat * H + n0 + t * 16 + ln)) * E + hf * 16;
#pragma unroll
    for (int c = 0; c < 4; ++c)
      acc[t] = wmma_bf16(A[c].v, *(const v16u*)(wr + c * 32), acc[t]);
  }

  const float* bias = (mat == 0) ? bq : (mat == 1) ? bk : bv;
  if (mat < 2) {
    unsigned short* dst = (mat == 0) ? qb : kb;
#pragma unroll
    for (int t = 0; t < 4; ++t) {
      const int h = n0 + t * 16 + ln;
      const float bv_ = bias[h];
#pragma unroll
      for (int r = 0; r < 8; ++r)
        dst[(size_t)(g0 + r + hf * 8) * H + h] = f2bf(acc[t][r] + bv_);
    }
  } else {
    const int bidx = g0 / T;
    const int t0 = (g0 % T) + hf * 8;      // 8 consecutive t per lane
#pragma unroll
    for (int t = 0; t < 4; ++t) {
      const int h = n0 + t * 16 + ln;
      const float bv_ = bias[h];
      v8u pk;
#pragma unroll
      for (int r = 0; r < 8; ++r) pk[r] = f2bf(acc[t][r] + bv_);
      *(v8u*)(vt + ((size_t)(bidx * H + h)) * T + t0) = pk;
    }
  }
}

// ---------------- flash attention (WMMA) ----------------
// One wave = 16 query rows of one batch. Q tile resident in VGPRs,
// stream causal key chunks of 32, online softmax, P*V accumulated fp32.

__global__ __launch_bounds__(32) void k_attn(const unsigned short* __restrict__ qb,
                                             const unsigned short* __restrict__ kb,
                                             const unsigned short* __restrict__ vt,
                                             float* __restrict__ out) {
  __shared__ __align__(64) unsigned short Plds[16 * 32];  // P transpose scratch
  const int lane = threadIdx.x;
  const int ln = lane & 15;
  const int hf = lane >> 4;
  const int q0 = blockIdx.x * 16;
  const int b = blockIdx.y;
  const float scale = 0.0625f;               // 1/sqrt(256)
  const float NEGINF = -__builtin_inff();

  // Q A-fragments for all 8 K-chunks of the head dim (resident: 64 VGPRs)
  U16x16 Q[8];
  {
    const unsigned short* qrow = qb + ((size_t)(b * T + q0 + ln)) * H;
#pragma unroll
    for (int c = 0; c < 8; ++c) {
      Q[c].h[0] = *(const v8u*)(qrow + c * 32 + hf * 16);
      Q[c].h[1] = *(const v8u*)(qrow + c * 32 + hf * 16 + 8);
    }
  }

  const v8f zero = {0.f, 0.f, 0.f, 0.f, 0.f, 0.f, 0.f, 0.f};
  v8f acc[16];                               // 16x256 fp32 output accumulator
#pragma unroll
  for (int t = 0; t < 16; ++t) acc[t] = zero;
  float mrow[8], lrow[8];                    // per-row max / sum (row = r + hf*8)
#pragma unroll
  for (int r = 0; r < 8; ++r) { mrow[r] = NEGINF; lrow[r] = 0.f; }

  const int nch = (q0 >> 5) + 1;             // causal: chunks of 32 keys
  for (int c = 0; c < nch; ++c) {
    const int s0 = c * 32;

    // ---- scores: S = Q * K^T  (two 16x16 tiles) ----
    v8f S0 = zero, S1 = zero;
    {
      const unsigned short* k0 = kb + ((size_t)(b * T + s0 + ln)) * H + hf * 16;
      const unsigned short* k1 = kb + ((size_t)(b * T + s0 + 16 + ln)) * H + hf * 16;
#pragma unroll
      for (int kc = 0; kc < 8; ++kc) {
        S0 = wmma_bf16(Q[kc].v, *(const v16u*)(k0 + kc * 32), S0);
        S1 = wmma_bf16(Q[kc].v, *(const v16u*)(k1 + kc * 32), S1);
      }
    }

    // ---- causal mask + online softmax (row stats across 16-lane halves) ----
    float p0[8], p1[8], csc[8];
#pragma unroll
    for (int r = 0; r < 8; ++r) {
      const int rowg = q0 + r + hf * 8;
      float a = (s0 + ln <= rowg) ? S0[r] * scale : NEGINF;
      float d = (s0 + 16 + ln <= rowg) ? S1[r] * scale : NEGINF;
      float t = fmaxf(a, d);
      t = fmaxf(t, __shfl_xor(t, 1));
      t = fmaxf(t, __shfl_xor(t, 2));
      t = fmaxf(t, __shfl_xor(t, 4));
      t = fmaxf(t, __shfl_xor(t, 8));
      const float mn = fmaxf(mrow[r], t);
      const float e0 = __expf(a - mn);
      const float e1 = __expf(d - mn);
      float rs = e0 + e1;
      rs += __shfl_xor(rs, 1);
      rs += __shfl_xor(rs, 2);
      rs += __shfl_xor(rs, 4);
      rs += __shfl_xor(rs, 8);
      const float sc = __expf(mrow[r] - mn);
      lrow[r] = lrow[r] * sc + rs;
      mrow[r] = mn;
      csc[r] = sc;
      p0[r] = e0;
      p1[r] = e1;
    }

#pragma unroll
    for (int t = 0; t < 16; ++t)
#pragma unroll
      for (int r = 0; r < 8; ++r) acc[t][r] *= csc[r];

    // ---- transpose P (D-layout -> A-layout) through LDS ----
    __syncthreads();
#pragma unroll
    for (int r = 0; r < 8; ++r) {
      Plds[(r + hf * 8) * 32 + ln] = f2bf(p0[r]);
      Plds[(r + hf * 8) * 32 + 16 + ln] = f2bf(p1[r]);
    }
    __syncthreads();

    U16x16 Pa;
    {
      const unsigned short* pr = Plds + ln * 32;
      Pa.h[0] = *(const v8u*)(pr + hf * 16);
      Pa.h[1] = *(const v8u*)(pr + hf * 16 + 8);
    }

    // ---- acc += P * V  (V^T layout: keys contiguous) ----
#pragma unroll
    for (int nt = 0; nt < 16; ++nt) {
      const unsigned short* vrow =
          vt + ((size_t)(b * H + nt * 16 + ln)) * T + s0 + hf * 16;
      acc[nt] = wmma_bf16(Pa.v, *(const v16u*)vrow, acc[nt]);
    }
  }

  // ---- epilogue: normalize and store fp32 ----
  float inv[8];
#pragma unroll
  for (int r = 0; r < 8; ++r) inv[r] = 1.0f / lrow[r];
#pragma unroll
  for (int nt = 0; nt < 16; ++nt) {
#pragma unroll
    for (int r = 0; r < 8; ++r) {
      out[((size_t)(b * T + q0 + r + hf * 8)) * H + nt * 16 + ln] =
          acc[nt][r] * inv[r];
    }
  }
}

// ---------------- host launcher ----------------

extern "C" void kernel_launch(void* const* d_in, const int* in_sizes, int n_in,
                              void* d_out, int out_size, void* d_ws, size_t ws_size,
                              hipStream_t stream) {
  (void)in_sizes; (void)n_in; (void)out_size; (void)ws_size;
  const float* x  = (const float*)d_in[0];
  const float* Wq = (const float*)d_in[1];
  const float* bq = (const float*)d_in[2];
  const float* Wk = (const float*)d_in[3];
  const float* bk = (const float*)d_in[4];
  const float* Wv = (const float*)d_in[5];
  const float* bv = (const float*)d_in[6];
  float* out = (float*)d_out;

  // workspace layout (bf16 as ushort): xb | wt | qb | kb | vt  (~28.2 MiB)
  char* ws = (char*)d_ws;
  unsigned short* xb = (unsigned short*)ws;
  unsigned short* wt = (unsigned short*)(ws + (size_t)M * E * 2);
  unsigned short* qb = (unsigned short*)(ws + (size_t)M * E * 2 + (size_t)3 * H * E * 2);
  unsigned short* kb = qb + (size_t)M * H;
  unsigned short* vt = kb + (size_t)M * H;

  k_cvt_x<<<(M * E + 255) / 256, 256, 0, stream>>>(x, xb, M * E);
  k_cvt_w<<<(3 * H * E + 255) / 256, 256, 0, stream>>>(Wq, Wk, Wv, wt);
  k_proj<<<dim3(M / 16, 12), 32, 0, stream>>>(xb, wt, bq, bk, bv, qb, kb, vt);
  k_attn<<<dim3(T / 16, B), 32, 0, stream>>>(qb, kb, vt, out);
}
